// OlyAttention_85693187490522
// MI455X (gfx1250) — compile-verified, matrix operand order checked by reading, not run on hardware
//
#include <hip/hip_runtime.h>
#include <hip/hip_bf16.h>

typedef __attribute__((ext_vector_type(16))) __bf16 v16bf;
typedef __attribute__((ext_vector_type(8)))  __bf16 v8bf;
typedef __attribute__((ext_vector_type(4)))  __bf16 v4bf;
typedef __attribute__((ext_vector_type(8)))  float  v8f;
typedef __attribute__((ext_vector_type(4)))  unsigned int u32x4;
typedef __attribute__((ext_vector_type(8)))  unsigned int u32x8;

#define CAT16(lo, hi) __builtin_shufflevector((lo), (hi), 0,1,2,3,4,5,6,7,8,9,10,11,12,13,14,15)

__device__ __forceinline__ v8f zero8() {
    v8f z;
#pragma unroll
    for (int i = 0; i < 8; ++i) z[i] = 0.0f;
    return z;
}

// ---------------------------------------------------------------------------
// TDM: DMA one 2D bf16 tile (rows x 32 elems) global -> LDS, padding each
// 32-element (16-DWORD) row with 4 DWORDs so LDS row stride = 40 bf16.
// D# group0: count=1, lds_addr, global_addr(57b), type=2.
// D# group1: data_size=2B, pad_enable, pad_interval=16dw, pad_amount=4dw,
//            tensor/tile dims, dim0 stride.
// ---------------------------------------------------------------------------
__device__ __forceinline__ void tdm_load_tile32(unsigned lds_addr,
                                                const __bf16* gaddr,
                                                unsigned rows,
                                                unsigned stride_elems) {
    const unsigned long long ga = (unsigned long long)(size_t)gaddr;
    u32x4 g0;
    g0[0] = 1u;                                               // count=1 (user)
    g0[1] = lds_addr;                                         // LDS byte addr
    g0[2] = (unsigned)ga;                                     // gaddr[31:0]
    g0[3] = ((unsigned)(ga >> 32) & 0x01FFFFFFu) | (2u << 30);// gaddr[56:32]|type=2
    u32x8 g1;
    g1[0] = (1u << 16)          // data_size = 2 bytes
          | (1u << 20)          // pad_enable
          | (3u << 22)          // pad_interval code 3 -> 16 DWORDs
          | (3u << 25);         // pad_amount  code 3 -> 4 DWORDs
    g1[1] = (32u & 0xFFFFu) << 16;                 // tensor_dim0 = 32 (lo16)
    g1[2] = (rows & 0xFFFFu) << 16;                // dim0 hi=0 | tensor_dim1 lo
    g1[3] = (32u << 16);                           // dim1 hi=0 | tile_dim0 = 32
    g1[4] = rows & 0xFFFFu;                        // tile_dim1 | tile_dim2=0
    g1[5] = stride_elems;                          // tensor_dim0_stride lo32
    g1[6] = 0u;
    g1[7] = 0u;
    asm volatile("tensor_load_to_lds %0, %1" :: "s"(g0), "s"(g1) : "memory");
}

// ---------------------------------------------------------------------------
// Elementwise f32 -> bf16 cast (4 elems/thread).
// ---------------------------------------------------------------------------
__global__ __launch_bounds__(256)
void cast_f32_bf16_kernel(const float* __restrict__ src, __bf16* __restrict__ dst) {
    const size_t i = ((size_t)blockIdx.x * 256 + threadIdx.x) * 4;
    const float4 f = *(const float4*)(src + i);
    v4bf b;
    b[0] = (__bf16)f.x; b[1] = (__bf16)f.y; b[2] = (__bf16)f.z; b[3] = (__bf16)f.w;
    *(v4bf*)(dst + i) = b;
}

// ---------------------------------------------------------------------------
// GEMM: C[M,N] = A[M,K] @ B[N,K]^T, bf16 WMMA, f32 accumulate.
// A/B tiles staged into padded LDS by the Tensor Data Mover (wave 0).
// MODE 0: epilogue scatters bf16 into q[b,h,s,d], k[b,h,s,d], vT[b,h,d,s].
// MODE 1: epilogue writes f32 to fout[M,N].
// Tile: 128(M) x 64(N) per workgroup, 8 waves, each wave 32x32 (2x2 WMMA).
// ---------------------------------------------------------------------------
template <int MODE>
__global__ __launch_bounds__(256)
void gemm_wmma_kernel(const __bf16* __restrict__ A,
                      const __bf16* __restrict__ Bw,
                      const int K,
                      __bf16* __restrict__ qout,
                      __bf16* __restrict__ kout,
                      __bf16* __restrict__ vTout,
                      float* __restrict__ fout,
                      const int Ncols) {
    __shared__ __bf16 As[128][40];   // 80B row stride (TDM pads rows)
    __shared__ __bf16 Bs[64][40];

    const int tid  = threadIdx.x;
    const int bm   = blockIdx.y;
    const int bn   = blockIdx.x;
    const int lane = tid & 31;
    const int wave = tid >> 5;
    const int wm   = wave & 3;      // 4 waves along M
    const int wn   = wave >> 2;     // 2 waves along N
    const int lm   = lane & 15;
    const int c0   = (lane & 16) ? 8 : 0;   // 16-bit A/B VGPR K-chunk select
    const int moff = (lane & 16) ? 8 : 0;   // C/D row offset for upper lanes

    const __bf16* Abase = A  + (size_t)(bm * 128) * K;
    const __bf16* Bbase = Bw + (size_t)(bn * 64) * K;
    const unsigned ldsA = (unsigned)(size_t)&As[0][0];
    const unsigned ldsB = (unsigned)(size_t)&Bs[0][0];

    // prefetch coordinates (A: one row per thread-pair, B: one row per 4 threads)
    const int arow = tid >> 1, acol = (tid & 1) * 16;
    const int brow = tid >> 2, bcol = (tid & 3) * 8;

    v8f acc[2][2];
#pragma unroll
    for (int i = 0; i < 2; ++i)
#pragma unroll
        for (int j = 0; j < 2; ++j) acc[i][j] = zero8();

    for (int k0 = 0; k0 < K; k0 += 32) {
        __syncthreads();                       // previous compute done
        if (wave == 0) {
            tdm_load_tile32(ldsA, Abase + k0, 128u, (unsigned)K);
            tdm_load_tile32(ldsB, Bbase + k0, 64u,  (unsigned)K);
            __builtin_amdgcn_s_wait_tensorcnt(0);
        }
        __syncthreads();                       // tiles visible to all waves

        if (k0 + 32 < K) {                     // L2 prefetch of next tiles
            __builtin_prefetch((const void*)(Abase + (size_t)arow * K + k0 + 32 + acol), 0, 1);
            __builtin_prefetch((const void*)(Bbase + (size_t)brow * K + k0 + 32 + bcol), 0, 1);
        }

        v16bf afr[2], bfr[2];
#pragma unroll
        for (int i = 0; i < 2; ++i) {
            const __bf16* p = &As[wm * 32 + i * 16 + lm][c0];
            v8bf lo = *(const v8bf*)p;
            v8bf hi = *(const v8bf*)(p + 16);
            afr[i] = CAT16(lo, hi);
        }
#pragma unroll
        for (int j = 0; j < 2; ++j) {
            const __bf16* p = &Bs[wn * 32 + j * 16 + lm][c0];
            v8bf lo = *(const v8bf*)p;
            v8bf hi = *(const v8bf*)(p + 16);
            bfr[j] = CAT16(lo, hi);
        }
#pragma unroll
        for (int i = 0; i < 2; ++i)
#pragma unroll
            for (int j = 0; j < 2; ++j)
                acc[i][j] = __builtin_amdgcn_wmma_f32_16x16x32_bf16(
                    false, afr[i], false, bfr[j], (short)0, acc[i][j], false, false);
    }

    // ---- epilogue ----
#pragma unroll
    for (int i = 0; i < 2; ++i)
#pragma unroll
        for (int j = 0; j < 2; ++j)
#pragma unroll
            for (int r = 0; r < 8; ++r) {
                const int m = bm * 128 + wm * 32 + i * 16 + r + moff;
                const int n = bn * 64 + wn * 32 + j * 16 + lm;
                const float v = acc[i][j][r];
                if (MODE == 0) {
                    const int which = n >> 11;          // 0=q 1=k 2=v
                    const int rem   = n & 2047;
                    const int hh    = rem >> 7;         // head
                    const int d     = rem & 127;
                    const int b     = m >> 11;
                    const int s     = m & 2047;
                    const int bh    = b * 16 + hh;
                    const __bf16 bv = (__bf16)v;
                    if (which == 0)
                        qout[((size_t)(bh * 2048 + s)) * 128 + d] = bv;
                    else if (which == 1)
                        kout[((size_t)(bh * 2048 + s)) * 128 + d] = bv;
                    else
                        vTout[((size_t)(bh * 128 + d)) * 2048 + s] = bv;
                } else {
                    fout[(size_t)m * Ncols + n] = v;
                }
            }
}

// ---------------------------------------------------------------------------
// In-place RoPE on bf16 q and k, one thread per (b,h,s,d-pair).
// ---------------------------------------------------------------------------
__global__ __launch_bounds__(256)
void rope_kernel(__bf16* __restrict__ q, __bf16* __restrict__ k,
                 const float* __restrict__ cosT, const float* __restrict__ sinT) {
    const int tid  = blockIdx.x * blockDim.x + threadIdx.x; // < 2*16*2048*64
    const int d    = tid & 63;
    const int s    = (tid >> 6) & 2047;
    const int rest = tid >> 17;                 // b*16 + h
    const size_t base = ((size_t)rest * 2048 + s) * 128;
    const float c  = cosT[s * 128 + d];
    const float sn = sinT[s * 128 + d];

    float q1 = (float)q[base + d], q2 = (float)q[base + d + 64];
    q[base + d]      = (__bf16)(q1 * c - q2 * sn);
    q[base + d + 64] = (__bf16)(q2 * c + q1 * sn);

    float k1 = (float)k[base + d], k2 = (float)k[base + d + 64];
    k[base + d]      = (__bf16)(k1 * c - k2 * sn);
    k[base + d + 64] = (__bf16)(k2 * c + k1 * sn);
}

// ---------------------------------------------------------------------------
// Flash-style causal attention. One wave per 16-query tile (4 waves/WG).
// 64-key blocks: QK^T = 16 WMMA, online softmax, PV = 16 WMMA per block.
// ---------------------------------------------------------------------------
__global__ __launch_bounds__(128)
void attn_kernel(const __bf16* __restrict__ Q,
                 const __bf16* __restrict__ Kt,
                 const __bf16* __restrict__ VT,
                 __bf16* __restrict__ ctx) {
    __shared__ __bf16 Pl[4][16][72];           // per-wave P staging, padded
    const int lane = threadIdx.x & 31;
    const int wave = threadIdx.x >> 5;
    const int flat = blockIdx.x * 4 + wave;    // b*16*128 + h*128 + qt
    const int qt   = flat & 127;
    const int h    = (flat >> 7) & 15;
    const int b    = flat >> 11;
    const int q0   = qt * 16;
    const int lm   = lane & 15;
    const int c0   = (lane & 16) ? 8 : 0;
    const int moff = (lane & 16) ? 8 : 0;

    const __bf16* qh = Q  + ((size_t)(b * 16 + h) * 2048) * 128;
    const __bf16* kh = Kt + ((size_t)(b * 16 + h) * 2048) * 128;
    const __bf16* vh = VT + ((size_t)(b * 16 + h) * 128) * 2048;

    v16bf qf[4];
    {
        const __bf16* qrow = qh + (size_t)(q0 + lm) * 128;
#pragma unroll
        for (int t = 0; t < 4; ++t) {
            v8bf lo = *(const v8bf*)(qrow + t * 32 + c0);
            v8bf hi = *(const v8bf*)(qrow + t * 32 + c0 + 16);
            qf[t] = CAT16(lo, hi);
        }
    }

    float rowM[8], rowL[8];
    v8f o[8];
#pragma unroll
    for (int r = 0; r < 8; ++r) { rowM[r] = -3e38f; rowL[r] = 0.0f; }
#pragma unroll
    for (int dt = 0; dt < 8; ++dt) o[dt] = zero8();

    __bf16* Pw = &Pl[wave][0][0];
    const int nkb = (q0 + 16 + 63) >> 6;       // causal: only blocks <= diag
    const float scale = 0.08838834764831845f;  // 1/sqrt(128)

    for (int kb = 0; kb < nkb; ++kb) {
        if (kb + 1 < nkb) {                    // L2 prefetch of next K/V block
            __builtin_prefetch((const void*)(kh + (size_t)((kb + 1) * 64 + lane * 2) * 128), 0, 1);
            __builtin_prefetch((const void*)(vh + (size_t)(lane * 4) * 2048 + (kb + 1) * 64), 0, 1);
        }
        // ---- S = Q @ K^T for 64 keys ----
        v8f st[4];
#pragma unroll
        for (int j = 0; j < 4; ++j) st[j] = zero8();
#pragma unroll
        for (int j = 0; j < 4; ++j) {
            const __bf16* krow = kh + (size_t)(kb * 64 + j * 16 + lm) * 128;
#pragma unroll
            for (int t = 0; t < 4; ++t) {
                v8bf lo = *(const v8bf*)(krow + t * 32 + c0);
                v8bf hi = *(const v8bf*)(krow + t * 32 + c0 + 16);
                v16bf kf = CAT16(lo, hi);
                st[j] = __builtin_amdgcn_wmma_f32_16x16x32_bf16(
                    false, qf[t], false, kf, (short)0, st[j], false, false);
            }
        }
        // ---- scale + causal mask + block row-max ----
        float nm[8];
#pragma unroll
        for (int r = 0; r < 8; ++r) {
            const int mrow = q0 + r + moff;
            float mv = -3e38f;
#pragma unroll
            for (int j = 0; j < 4; ++j) {
                const int kcol = kb * 64 + j * 16 + lm;
                float v = st[j][r] * scale;
                v = (kcol <= mrow) ? v : -1e9f;
                st[j][r] = v;
                mv = fmaxf(mv, v);
            }
            mv = fmaxf(mv, __shfl_xor(mv, 1, 32));
            mv = fmaxf(mv, __shfl_xor(mv, 2, 32));
            mv = fmaxf(mv, __shfl_xor(mv, 4, 32));
            mv = fmaxf(mv, __shfl_xor(mv, 8, 32));
            nm[r] = fmaxf(rowM[r], mv);
        }
        // ---- rescale running state, exp, stage P (bf16) into LDS ----
#pragma unroll
        for (int r = 0; r < 8; ++r) {
            const float alpha = __expf(rowM[r] - nm[r]);
            rowM[r] = nm[r];
            rowL[r] *= alpha;
#pragma unroll
            for (int dt = 0; dt < 8; ++dt) o[dt][r] *= alpha;
            float rs = 0.0f;
#pragma unroll
            for (int j = 0; j < 4; ++j) {
                const float p = __expf(st[j][r] - nm[r]);
                rs += p;
                Pw[(r + moff) * 72 + j * 16 + lm] = (__bf16)p;
            }
            rs += __shfl_xor(rs, 1, 32);
            rs += __shfl_xor(rs, 2, 32);
            rs += __shfl_xor(rs, 4, 32);
            rs += __shfl_xor(rs, 8, 32);
            rowL[r] += rs;
        }
        // ---- O += P @ V  (V stored transposed: vT[d][s]) ----
#pragma unroll
        for (int t = 0; t < 2; ++t) {
            v8bf plo = *(const v8bf*)(Pw + lm * 72 + t * 32 + c0);
            v8bf phi = *(const v8bf*)(Pw + lm * 72 + t * 32 + c0 + 16);
            v16bf pf = CAT16(plo, phi);
#pragma unroll
            for (int dt = 0; dt < 8; ++dt) {
                const __bf16* vrow = vh + (size_t)(dt * 16 + lm) * 2048 + kb * 64 + t * 32;
                v8bf vlo = *(const v8bf*)(vrow + c0);
                v8bf vhi = *(const v8bf*)(vrow + c0 + 16);
                v16bf vf = CAT16(vlo, vhi);
                o[dt] = __builtin_amdgcn_wmma_f32_16x16x32_bf16(
                    false, pf, false, vf, (short)0, o[dt], false, false);
            }
        }
    }

    // ---- normalize and write ctx[b*S + m][h*128 + d] as bf16 ----
#pragma unroll
    for (int dt = 0; dt < 8; ++dt)
#pragma unroll
        for (int r = 0; r < 8; ++r) {
            const float vv = o[dt][r] / rowL[r];
            const int m = q0 + r + moff;
            ctx[((size_t)(b * 2048 + m)) * 2048 + h * 128 + dt * 16 + lm] = (__bf16)vv;
        }
}

// ---------------------------------------------------------------------------
extern "C" void kernel_launch(void* const* d_in, const int* in_sizes, int n_in,
                              void* d_out, int out_size, void* d_ws, size_t ws_size,
                              hipStream_t stream) {
    (void)in_sizes; (void)n_in; (void)out_size; (void)ws_size;
    const float* hidden = (const float*)d_in[0];
    const float* cosT   = (const float*)d_in[1];
    const float* sinT   = (const float*)d_in[2];
    // d_in[3] attention_mask: causal mask applied analytically in-kernel
    const float* wqkv   = (const float*)d_in[4];
    const float* wo     = (const float*)d_in[5];
    float* out = (float*)d_out;

    const size_t SZ_H  = (size_t)4096 * 2048 * 2;           // hidden  bf16
    const size_t SZ_W1 = (size_t)6144 * 2048 * 2;           // w_qkv   bf16
    const size_t SZ_W2 = (size_t)2048 * 2048 * 2;           // w_o     bf16
    const size_t SZ_T  = (size_t)2 * 16 * 2048 * 128 * 2;   // q/k/vT/ctx each

    char* ws = (char*)d_ws;
    __bf16* hb   = (__bf16*)(ws);
    __bf16* wqb  = (__bf16*)(ws + SZ_H);
    __bf16* wob  = (__bf16*)(ws + SZ_H + SZ_W1);
    __bf16* qb   = (__bf16*)(ws + SZ_H + SZ_W1 + SZ_W2);
    __bf16* kbuf = (__bf16*)(ws + SZ_H + SZ_W1 + SZ_W2 + SZ_T);
    __bf16* vT   = (__bf16*)(ws + SZ_H + SZ_W1 + SZ_W2 + 2 * SZ_T);
    __bf16* ctx  = (__bf16*)(ws + SZ_H + SZ_W1 + SZ_W2 + 3 * SZ_T);

    // 0) one-pass f32 -> bf16 casts (enables TDM byte-copy staging in GEMMs)
    cast_f32_bf16_kernel<<<(4096 * 2048 / 4) / 256, 256, 0, stream>>>(hidden, hb);
    cast_f32_bf16_kernel<<<(6144 * 2048 / 4) / 256, 256, 0, stream>>>(wqkv, wqb);
    cast_f32_bf16_kernel<<<(2048 * 2048 / 4) / 256, 256, 0, stream>>>(wo, wob);

    // 1) QKV projection (TDM-staged, scatter into q/k/vT bf16)
    gemm_wmma_kernel<0><<<dim3(6144 / 64, 4096 / 128), 256, 0, stream>>>(
        hb, wqb, 2048, qb, kbuf, vT, nullptr, 0);
    // 2) RoPE on q and k (in place)
    rope_kernel<<<(2 * 16 * 2048 * 64) / 256, 256, 0, stream>>>(qb, kbuf, cosT, sinT);
    // 3) Causal flash attention -> ctx (bf16)
    attn_kernel<<<(2 * 16 * 128) / 4, 128, 0, stream>>>(qb, kbuf, vT, ctx);
    // 4) Output projection (TDM-staged) -> f32 d_out
    gemm_wmma_kernel<1><<<dim3(2048 / 64, 4096 / 128), 256, 0, stream>>>(
        ctx, wob, 2048, nullptr, nullptr, nullptr, out, 2048);
}
